// RNNLayer_87600152969279
// MI455X (gfx1250) — matrix operand order repaired and checked by reading, not compiled, not run on hardware
//
#include <hip/hip_runtime.h>
#include <hip/hip_bf16.h>

// ---------------------------------------------------------------------------
// GRU layer, B=128, S=1024, D=H=256 (3H = 768 gate columns), fp32 in/out.
// Persistent batch-partitioned scan: 8 workgroups x 16 batch rows, 512 threads
// (16 wave32), h resident in LDS for all 1024 steps. Matrix math in bf16 WMMA
// (v_wmma_f32_16x16x32_bf16), f32 accumulate. Weights pre-packed into WMMA
// B-fragment order (bf16) in workspace, streamed from L2 every step via
// GLOBAL (address_space(1)) loads off a laundered SGPR base.
// ---------------------------------------------------------------------------

#define B_  128
#define S_  1024
#define D_  256
#define H_  256
#define G3_ 768           // 3*H, gate columns (r | z | n)
#define KT_ 8             // K tiles of 32 over K=256
#define CT_ 48            // column tiles of 16 over 768
#define ROWS_ 16          // batch rows per workgroup
#define THREADS_ 512      // 16 waves
#define WAVES_ 16
#define CT_PER_WAVE_ 3    // 48 / 16

#define GAS __attribute__((address_space(1)))

typedef __attribute__((ext_vector_type(16))) __bf16 v16bf;
typedef __attribute__((ext_vector_type(8)))  __bf16 v8bf;
typedef __attribute__((ext_vector_type(8)))  float  v8f;

// packed fragment blob layout: frag[ct][ks][lane][16] bf16, 1 KB per fragment
#define FRAG_ELEMS_ (32 * 16)

// ---------------------------------------------------------------------------
// Pack W (K=256 x 768, row-major f32) into bf16 WMMA B-fragment order.
// B operand of v_wmma_f32_16x16x32_bf16 is a 32x16 (K x N) tile:
//   lane L -> column N = L&15, K half = L>>4; 16 bf16 values per lane.
// ---------------------------------------------------------------------------
__global__ void gru_pack_weights(const float* __restrict__ Wih,
                                 const float* __restrict__ Whh,
                                 __bf16* __restrict__ pIh,
                                 __bf16* __restrict__ pHh) {
    int tid = blockIdx.x * blockDim.x + threadIdx.x;   // 48*8*32 = 12288 threads
    if (tid >= CT_ * KT_ * 32) return;
    int lane = tid & 31;
    int ks   = (tid >> 5) & 7;
    int ct   = tid >> 8;
    int n    = lane & 15;
    int half = lane >> 4;
    int col  = ct * 16 + n;
    __bf16* oI = pIh + tid * 16;
    __bf16* oH = pHh + tid * 16;
#pragma unroll
    for (int j = 0; j < 16; ++j) {
        int k = ks * 32 + half * 8 + (j & 7) + (j >> 3) * 16;
        oI[j] = (__bf16)Wih[k * G3_ + col];
        oH[j] = (__bf16)Whh[k * G3_ + col];
    }
}

// Load a 16x32 bf16 A fragment (rows = batch rows, K slice ks) from a
// row-major bf16 [16][256] LDS tile. Lane L (m = L&15) holds two contiguous
// 8-element K chunks: [h*8, h*8+8) and [16+h*8, ...), h = L>>4, + ks*32.
__device__ __forceinline__ v16bf load_a_frag(const __bf16* __restrict__ lds,
                                             int lane, int ks) {
    int m    = lane & 15;
    int half = lane >> 4;
    int base = m * 256 + ks * 32 + half * 8;
    v8bf lo = *(const v8bf*)(lds + base);
    v8bf hi = *(const v8bf*)(lds + base + 16);
    v16bf a;
#pragma unroll
    for (int i = 0; i < 8; ++i) { a[i] = lo[i]; a[i + 8] = hi[i]; }
    return a;
}

__device__ __forceinline__ float sigmoid_f(float x) {
    return 1.0f / (1.0f + __expf(-x));
}

// ---------------------------------------------------------------------------
// Persistent scan kernel: one workgroup per 16 batch rows, loops t = 0..S-1.
// ---------------------------------------------------------------------------
__global__ __launch_bounds__(THREADS_, 1)
void gru_scan_kernel(const float* __restrict__ x,      // [B,S,D]
                     const float* __restrict__ mask,   // [B,S]
                     const float* __restrict__ b_ih,   // [768]
                     const float* __restrict__ b_hh,   // [768]
                     const __bf16* __restrict__ pIh,   // packed fragments
                     const __bf16* __restrict__ pHh,   // packed fragments
                     float* __restrict__ y)            // [B,S,H]
{
    extern __shared__ __align__(16) char smem[];
    float*  gxs = (float*)smem;                 // [16][768]  48 KB
    float*  ghs = gxs + ROWS_ * G3_;            // [16][768]  48 KB
    float*  hF  = ghs + ROWS_ * G3_;            // [16][256]  16 KB (f32 state)
    float*  bI  = hF + ROWS_ * H_;              // [768]       3 KB (bias ih)
    float*  bH  = bI + G3_;                     // [768]       3 KB (bias hh)
    __bf16* ha  = (__bf16*)(bH + G3_);          // [16][256]   8 KB (bf16 A form)
    __bf16* xa  = ha + ROWS_ * H_;              // [16][256]   8 KB (bf16 A form)

    const int tid  = threadIdx.x;
    const int lane = tid & 31;
    const int wave = tid >> 5;
    const int b0   = blockIdx.x * ROWS_;

    // elementwise mapping: each thread owns 8 consecutive columns of one row
    const int er = tid >> 5;          // row 0..15
    const int ec = (tid & 31) * 8;    // col 0..248

    // init h = 0 and stage biases into LDS (keeps them out of VGPRs)
#pragma unroll
    for (int e = 0; e < 8; ++e) {
        hF[er * H_ + ec + e] = 0.0f;
        ha[er * H_ + ec + e] = (__bf16)0.0f;
    }
    for (int i = tid; i < G3_; i += THREADS_) {
        bI[i] = b_ih[i];
        bH[i] = b_hh[i];
    }
    __syncthreads();

#pragma unroll 1
    for (int t = 0; t < S_; ++t) {
        // Launder the weight-blob bases as 64-bit integers held in SGPRs
        // (loop-carried: defeats LICM hoist/spill), then cast to explicit
        // GLOBAL address-space pointers so the loads lower as
        // global_load_b128 (LOADcnt only, saddr+imm form), not flat_load.
        unsigned long long wIh = (unsigned long long)pIh;
        unsigned long long wHh = (unsigned long long)pHh;
        asm volatile("" : "+s"(wIh), "+s"(wHh));
        const GAS __bf16* gIh = (const GAS __bf16*)wIh;
        const GAS __bf16* gHh = (const GAS __bf16*)wHh;

        // ---- phase 1: stage x_t tile [16 x 256] f32 -> bf16 into LDS ------
        {
            const float* xr = x + ((size_t)(b0 + er) * S_ + t) * D_ + ec;
            float4 a0 = *(const float4*)(xr);
            float4 a1 = *(const float4*)(xr + 4);
            __bf16* dst = xa + er * D_ + ec;
            dst[0] = (__bf16)a0.x; dst[1] = (__bf16)a0.y;
            dst[2] = (__bf16)a0.z; dst[3] = (__bf16)a0.w;
            dst[4] = (__bf16)a1.x; dst[5] = (__bf16)a1.y;
            dst[6] = (__bf16)a1.z; dst[7] = (__bf16)a1.w;
            if (t + 1 < S_)
                __builtin_prefetch(xr + D_, 0, 1);   // global_prefetch_b8
        }
        __syncthreads();

        // ---- phase 2: gx = x_t @ W_ih, gh = h @ W_hh via WMMA -------------
        v8f accx[CT_PER_WAVE_];
        v8f acch[CT_PER_WAVE_];
#pragma unroll
        for (int j = 0; j < CT_PER_WAVE_; ++j) { accx[j] = (v8f)0.0f; acch[j] = (v8f)0.0f; }

#pragma unroll
        for (int ks = 0; ks < KT_; ++ks) {
            v16bf axf = load_a_frag(xa, lane, ks);
            v16bf ahf = load_a_frag(ha, lane, ks);
#pragma unroll
            for (int j = 0; j < CT_PER_WAVE_; ++j) {
                int ct = wave + j * WAVES_;
                unsigned off = (unsigned)(ct * KT_ + ks) * FRAG_ELEMS_ + (unsigned)lane * 16;
                v16bf bi = *(const GAS v16bf*)(gIh + off);
                v16bf bh = *(const GAS v16bf*)(gHh + off);
                accx[j] = __builtin_amdgcn_wmma_f32_16x16x32_bf16(
                    false, axf, false, bi, (short)0, accx[j], false, false);
                acch[j] = __builtin_amdgcn_wmma_f32_16x16x32_bf16(
                    false, ahf, false, bh, (short)0, acch[j], false, false);
            }
        }

        // spill accumulators to LDS gate buffers.
        // C layout: VGPR i -> row m = i + (lane>>4)*8, col n = lane&15.
        {
            int n    = lane & 15;
            int rb   = (lane >> 4) * 8;
#pragma unroll
            for (int j = 0; j < CT_PER_WAVE_; ++j) {
                int cb = (wave + j * WAVES_) * 16 + n;
#pragma unroll
                for (int i = 0; i < 8; ++i) {
                    gxs[(rb + i) * G3_ + cb] = accx[j][i];
                    ghs[(rb + i) * G3_ + cb] = acch[j][i];
                }
            }
        }
        __syncthreads();

        // ---- phase 3: gates + state update + output store -----------------
        {
            const float mval = mask[(size_t)(b0 + er) * S_ + t];
            float* yout = y + ((size_t)(b0 + er) * S_ + t) * H_ + ec;
#pragma unroll
            for (int e = 0; e < 8; ++e) {
                int c = ec + e;
                float gxr = gxs[er * G3_ + c]        + bI[c];
                float ghr = ghs[er * G3_ + c]        + bH[c];
                float gxz = gxs[er * G3_ + c + H_]   + bI[c + H_];
                float ghz = ghs[er * G3_ + c + H_]   + bH[c + H_];
                float gxn = gxs[er * G3_ + c + 2*H_] + bI[c + 2*H_];
                float ghn = ghs[er * G3_ + c + 2*H_] + bH[c + 2*H_];
                float r  = sigmoid_f(gxr + ghr);
                float z  = sigmoid_f(gxz + ghz);
                float n  = tanhf(gxn + r * ghn);
                float hp = hF[er * H_ + c];
                float hn = (1.0f - z) * n + z * hp;
                float ho = mval * hn + (1.0f - mval) * hp;
                hF[er * H_ + c] = ho;
                ha[er * H_ + c] = (__bf16)ho;
                yout[e] = ho;
            }
        }
        __syncthreads();
    }
}

extern "C" void kernel_launch(void* const* d_in, const int* in_sizes, int n_in,
                              void* d_out, int out_size, void* d_ws, size_t ws_size,
                              hipStream_t stream) {
    const float* x    = (const float*)d_in[0];
    const float* mask = (const float*)d_in[1];
    const float* Wih  = (const float*)d_in[2];
    const float* Whh  = (const float*)d_in[3];
    const float* bih  = (const float*)d_in[4];
    const float* bhh  = (const float*)d_in[5];
    float* y = (float*)d_out;

    // workspace: packed bf16 weight fragments, 48*8 frags * 512 bf16 each
    const size_t packElems = (size_t)CT_ * KT_ * FRAG_ELEMS_;   // 196608
    __bf16* pIh = (__bf16*)d_ws;
    __bf16* pHh = pIh + packElems;

    // 1) pack weights into WMMA B-fragment order (bf16)
    {
        int total = CT_ * KT_ * 32;           // 12288 threads
        gru_pack_weights<<<(total + 255) / 256, 256, 0, stream>>>(Wih, Whh, pIh, pHh);
    }

    // 2) persistent batch-partitioned scan: 8 WGs x 512 threads, ~134 KB LDS
    {
        size_t smem = (size_t)(ROWS_ * G3_ * 4) * 2   // gxs + ghs
                    + (size_t)(ROWS_ * H_ * 4)        // hF
                    + (size_t)(G3_ * 4) * 2           // bI + bH
                    + (size_t)(ROWS_ * H_ * 2) * 2;   // ha + xa
        gru_scan_kernel<<<B_ / ROWS_, THREADS_, smem, stream>>>(
            x, mask, bih, bhh, pIh, pHh, y);
    }
}
